// IDGGNNBlock_77592879170078
// MI455X (gfx1250) — matrix operand order repaired and checked by reading, not compiled, not run on hardware
//
#include <hip/hip_runtime.h>
#include <hip/hip_bf16.h>

// ---------------------------------------------------------------------------
// IDG GNN block for MI455X (gfx1250, wave32, WMMA).
// bf16 activations/weights feed v_wmma_f32_16x16x32_bf16 with f32 accumulate;
// fragments are contiguous 16B/32B vector loads. Edge stage: fused
// el-projection from LDS + sigmoid gate + atomic scatter-add (L2-resident).
// ---------------------------------------------------------------------------

typedef __attribute__((ext_vector_type(16))) __bf16 v16bf;
typedef __attribute__((ext_vector_type(8)))  __bf16 v8bf;
typedef __attribute__((ext_vector_type(4)))  __bf16 v4bf;
typedef __attribute__((ext_vector_type(8)))  float  v8f;

#define HDIM   128
#define EHDIM  32
#define LAYERS 3

__device__ __forceinline__ float sigmoidf(float x) {
    return 1.0f / (1.0f + __expf(-x));
}

__device__ __forceinline__ void storeOut(float* p, float v)  { *p = v; }
__device__ __forceinline__ void storeOut(__bf16* p, float v) { *p = (__bf16)v; }

// ---------------------------------------------------------------------------
// GEMM v1 (fp32 A via float4 loads, fp32 W, convert on the fly). Used only
// for the two input projections whose A operands arrive as fp32 and are read
// once. One wave = one 16x16 tile. M%16==0, Nout%16==0, K%32==0.
// ---------------------------------------------------------------------------
template <typename OutT>
__global__ __launch_bounds__(256) void wmma_gemm_f32a(
    const float* __restrict__ A, const float* __restrict__ W,
    const float* __restrict__ bias, OutT* __restrict__ C,
    int M, int K, int Nout)
{
    const int lane = threadIdx.x & 31;
    const int wave = threadIdx.x >> 5;
    const int ntc  = Nout >> 4;
    const int T    = (M >> 4) * ntc;
    const int t    = blockIdx.x * 8 + wave;
    if (t >= T) return;

    const int m0   = (t / ntc) << 4;
    const int n0   = (t % ntc) << 4;
    const int half = lane >> 4;
    const int l15  = lane & 15;
    const int m    = m0 + l15;
    const int n    = n0 + l15;
    const int kbA  = half * 16;
    const int kbB  = half * 16;

    v8f acc = {};
    for (int k0 = 0; k0 < K; k0 += 32) {
        // A fragment: one contiguous 64B fp32 run -> 4 x b128 loads
        const float* arow = A + (size_t)m * K + (k0 + kbA);
        const float4 a0 = *(const float4*)(arow + 0);
        const float4 a1 = *(const float4*)(arow + 4);
        const float4 a2 = *(const float4*)(arow + 8);
        const float4 a3 = *(const float4*)(arow + 12);
        v16bf a, b;
        a[0]  = (__bf16)a0.x; a[1]  = (__bf16)a0.y;
        a[2]  = (__bf16)a0.z; a[3]  = (__bf16)a0.w;
        a[4]  = (__bf16)a1.x; a[5]  = (__bf16)a1.y;
        a[6]  = (__bf16)a1.z; a[7]  = (__bf16)a1.w;
        a[8]  = (__bf16)a2.x; a[9]  = (__bf16)a2.y;
        a[10] = (__bf16)a2.z; a[11] = (__bf16)a2.w;
        a[12] = (__bf16)a3.x; a[13] = (__bf16)a3.y;
        a[14] = (__bf16)a3.z; a[15] = (__bf16)a3.w;

        // B fragment: strided column reads (tiny weight, L0/L2 resident)
        const float* wp = W + (size_t)(k0 + kbB) * Nout + n;
#pragma unroll
        for (int j = 0; j < 16; ++j) b[j] = (__bf16)wp[(size_t)j * Nout];

        acc = __builtin_amdgcn_wmma_f32_16x16x32_bf16(
            false, a, false, b, (short)0, acc, false, false);
    }
    const float bval  = bias[n];
    const int   mBase = m0 + half * 8;
#pragma unroll
    for (int r = 0; r < 8; ++r)
        storeOut(C + (size_t)(mBase + r) * Nout + n, acc[r] + bval);
}

// ---------------------------------------------------------------------------
// GEMM v2 (bf16 A, pre-transposed bf16 Wt[Nout][K]).
// Each wave owns one 16-column tile, preloads ALL B fragments into registers
// (KSTEPS*8 VGPRs), then grid-strides over M tiles: steady state is two 16B
// A loads + one WMMA per K-step. Requires gridWaves % (Nout/16) == 0.
// ---------------------------------------------------------------------------
template <int KSTEPS, typename OutT>
__global__ __launch_bounds__(256) void wmma_gemm_bf16a(
    const __bf16* __restrict__ A, const __bf16* __restrict__ Wt,
    const float* __restrict__ bias, OutT* __restrict__ C,
    int M, int Nout)
{
    constexpr int K = KSTEPS * 32;
    const int lane = threadIdx.x & 31;
    const int wave = threadIdx.x >> 5;
    const int ntc  = Nout >> 4;
    const int nw   = gridDim.x * 8;
    const int t    = blockIdx.x * 8 + wave;

    const int colT      = t % ntc;
    const int rowStride = nw / ntc;
    const int mtiles    = M >> 4;
    const int half      = lane >> 4;
    const int l15       = lane & 15;
    const int n         = colT * 16 + l15;
    const int kbA       = half * 16;
    const int kbB       = half * 16;

    // Preload all B fragments (weights stay in registers across M tiles)
    v16bf bfrag[KSTEPS];
    const __bf16* wrow = Wt + (size_t)n * K;
#pragma unroll
    for (int s = 0; s < KSTEPS; ++s)
        bfrag[s] = *(const v16bf*)(wrow + s * 32 + kbB);

    const float bval = bias[n];

    for (int mt = t / ntc; mt < mtiles; mt += rowStride) {
        const int m0 = mt << 4;
        const __bf16* arow = A + (size_t)(m0 + l15) * K + kbA;
        v8f acc = {};
#pragma unroll
        for (int s = 0; s < KSTEPS; ++s) {
            const v8bf alo = *(const v8bf*)(arow + s * 32);
            const v8bf ahi = *(const v8bf*)(arow + s * 32 + 8);
            const v16bf a  = __builtin_shufflevector(
                alo, ahi, 0, 1, 2, 3, 4, 5, 6, 7, 8, 9, 10, 11, 12, 13, 14, 15);
            acc = __builtin_amdgcn_wmma_f32_16x16x32_bf16(
                false, a, false, bfrag[s], (short)0, acc, false, false);
        }
        const int mBase = m0 + half * 8;
#pragma unroll
        for (int r = 0; r < 8; ++r)
            storeOut(C + (size_t)(mBase + r) * Nout + n, acc[r] + bval);
    }
}

// ---------------------------------------------------------------------------
// Weight prep: Wt[n*K + k] = (bf16) W[k*Nout + n]   (transpose + cast, tiny)
// ---------------------------------------------------------------------------
__global__ __launch_bounds__(256) void prep_w(
    const float* __restrict__ W, __bf16* __restrict__ Wt, int K, int Nout)
{
    const int idx = blockIdx.x * 256 + threadIdx.x;
    if (idx < K * Nout) {
        const int k = idx / Nout;
        const int n = idx % Nout;
        Wt[(size_t)n * K + k] = (__bf16)W[idx];
    }
}

// ---------------------------------------------------------------------------
// Edge kernel: per edge e with endpoints (s=src, d=dst):
//   el   = eproj[e] @ Wed + bed          (fused 32x128 mini-GEMM from LDS)
//   gate = sigmoid(k[d] + q[s] + el)
//   agg[d] += gate * v[s]                (atomic scatter-add, L2-resident)
// One wave per edge (grid-stride); each lane owns 4 of 128 channels.
// ---------------------------------------------------------------------------
__global__ __launch_bounds__(256) void edge_gate_agg(
    const float* __restrict__ kf, const float* __restrict__ qf,
    const float* __restrict__ vf, const __bf16* __restrict__ eproj,
    const int* __restrict__ src, const int* __restrict__ dst,
    const float* __restrict__ Wed, const float* __restrict__ bed,
    float* __restrict__ agg, int E)
{
    __shared__ float sW[EHDIM * HDIM];   // 16 KB of the 320 KB WGP LDS
    __shared__ float sB[HDIM];
    for (int i = threadIdx.x; i < EHDIM * HDIM; i += 256) sW[i] = Wed[i];
    for (int i = threadIdx.x; i < HDIM; i += 256)         sB[i] = bed[i];
    __syncthreads();

    const int lane = threadIdx.x & 31;
    const int gw   = blockIdx.x * 8 + (threadIdx.x >> 5);
    const int nw   = gridDim.x * 8;
    const int c0   = lane * 4;

    for (int e = gw; e < E; e += nw) {
        if (e + nw < E)  // gfx1250 global_prefetch for the next edge's e-row
            __builtin_prefetch(eproj + (size_t)(e + nw) * EHDIM, 0, 1);

        const int s = src[e];
        const int d = dst[e];
        const float ev = (float)eproj[(size_t)e * EHDIM + lane];

        float4 el = *(const float4*)(sB + c0);
#pragma unroll
        for (int j = 0; j < EHDIM; ++j) {
            const float  ej = __shfl(ev, j, 32);
            const float4 w  = *(const float4*)(sW + j * HDIM + c0);
            el.x += ej * w.x; el.y += ej * w.y;
            el.z += ej * w.z; el.w += ej * w.w;
        }

        const float4 kd = *(const float4*)(kf + (size_t)d * HDIM + c0);
        const float4 qs = *(const float4*)(qf + (size_t)s * HDIM + c0);
        const float4 vs = *(const float4*)(vf + (size_t)s * HDIM + c0);

        float4 msg;
        msg.x = sigmoidf(kd.x + qs.x + el.x) * vs.x;
        msg.y = sigmoidf(kd.y + qs.y + el.y) * vs.y;
        msg.z = sigmoidf(kd.z + qs.z + el.z) * vs.z;
        msg.w = sigmoidf(kd.w + qs.w + el.w) * vs.w;

        float* ap = agg + (size_t)d * HDIM + c0;
        atomicAdd(ap + 0, msg.x);
        atomicAdd(ap + 1, msg.y);
        atomicAdd(ap + 2, msg.z);
        atomicAdd(ap + 3, msg.w);
    }
}

// ---------------------------------------------------------------------------
// h = LayerNorm(agg + xskip) * g + b ; write bf16 h into x (next layer GEMM
// input) and the concat buffer. One wave per node row, wave32 shuffles.
// ---------------------------------------------------------------------------
__global__ __launch_bounds__(256) void ln_res(
    const float* __restrict__ agg, const float* __restrict__ xskip,
    const float* __restrict__ g, const float* __restrict__ b,
    __bf16* __restrict__ xout, __bf16* __restrict__ cat, int Nn, int layer)
{
    const int lane = threadIdx.x & 31;
    const int node = blockIdx.x * 8 + (threadIdx.x >> 5);
    if (node >= Nn) return;
    const int c0 = lane * 4;

    const float4 a4 = *(const float4*)(agg   + (size_t)node * HDIM + c0);
    const float4 s4 = *(const float4*)(xskip + (size_t)node * HDIM + c0);
    float4 h;
    h.x = a4.x + s4.x; h.y = a4.y + s4.y;
    h.z = a4.z + s4.z; h.w = a4.w + s4.w;

    float sum = h.x + h.y + h.z + h.w;
    float sq  = h.x * h.x + h.y * h.y + h.z * h.z + h.w * h.w;
#pragma unroll
    for (int o = 16; o > 0; o >>= 1) {
        sum += __shfl_xor(sum, o, 32);
        sq  += __shfl_xor(sq,  o, 32);
    }
    const float mean = sum * (1.0f / HDIM);
    const float var  = sq  * (1.0f / HDIM) - mean * mean;
    const float rstd = rsqrtf(var + 1e-5f);

    const float4 g4 = *(const float4*)(g + c0);
    const float4 b4 = *(const float4*)(b + c0);
    v4bf o4;
    o4[0] = (__bf16)((h.x - mean) * rstd * g4.x + b4.x);
    o4[1] = (__bf16)((h.y - mean) * rstd * g4.y + b4.y);
    o4[2] = (__bf16)((h.z - mean) * rstd * g4.z + b4.z);
    o4[3] = (__bf16)((h.w - mean) * rstd * g4.w + b4.w);

    *(v4bf*)(xout + (size_t)node * HDIM + c0) = o4;
    *(v4bf*)(cat + (size_t)node * (LAYERS * HDIM) + layer * HDIM + c0) = o4;
}

__global__ __launch_bounds__(256) void zero_f32(float* __restrict__ p, size_t n) {
    size_t i = (size_t)blockIdx.x * 256 + threadIdx.x;
    if (i < n) p[i] = 0.0f;
}

// ---------------------------------------------------------------------------
extern "C" void kernel_launch(void* const* d_in, const int* in_sizes, int n_in,
                              void* d_out, int out_size, void* d_ws, size_t ws_size,
                              hipStream_t stream)
{
    const float* nf  = (const float*)d_in[0];
    const int*   ei  = (const int*  )d_in[1];
    const float* ea  = (const float*)d_in[2];
    const float* Wn  = (const float*)d_in[3];
    const float* bn  = (const float*)d_in[4];
    const float* We  = (const float*)d_in[5];
    const float* be  = (const float*)d_in[6];
    const float* Wk  = (const float*)d_in[7];
    const float* bk  = (const float*)d_in[8];
    const float* Wq  = (const float*)d_in[9];
    const float* bq  = (const float*)d_in[10];
    const float* Wv  = (const float*)d_in[11];
    const float* bv  = (const float*)d_in[12];
    const float* Wed = (const float*)d_in[13];
    const float* bed = (const float*)d_in[14];
    const float* Wsk = (const float*)d_in[15];
    const float* cb  = (const float*)d_in[16];
    const float* lng = (const float*)d_in[17];
    const float* lnb = (const float*)d_in[18];
    const float* Wh  = (const float*)d_in[19];
    const float* bh  = (const float*)d_in[20];
    (void)n_in; (void)out_size; (void)ws_size;

    const int IN_NODE = 256, IN_EDGE = 64;
    const int Nn = in_sizes[0] / IN_NODE;   // 50000
    const int E  = in_sizes[1] / 2;         // 1600000
    const int* src = ei;
    const int* dst = ei + E;

    // ---- workspace layout (fp32 blocks first, then bf16; 64B aligned) ----
    char* wsb = (char*)d_ws;
    size_t off = 0;
    auto alloc = [&](size_t bytes) {
        void* p = wsb + off;
        off += (bytes + 63) & ~(size_t)63;
        return p;
    };
    const size_t nhf = (size_t)Nn * HDIM;
    float* kf   = (float*)alloc(nhf * 4);
    float* qf   = (float*)alloc(nhf * 4);
    float* vf   = (float*)alloc(nhf * 4);
    float* xs   = (float*)alloc(nhf * 4);
    float* agg  = (float*)alloc(nhf * 4);
    __bf16* xb  = (__bf16*)alloc(nhf * 2);
    __bf16* epb = (__bf16*)alloc((size_t)E * EHDIM * 2);
    __bf16* ctb = (__bf16*)alloc(nhf * LAYERS * 2);
    __bf16* WtK = (__bf16*)alloc((size_t)LAYERS * HDIM * HDIM * 2);
    __bf16* WtQ = (__bf16*)alloc((size_t)LAYERS * HDIM * HDIM * 2);
    __bf16* WtV = (__bf16*)alloc((size_t)LAYERS * HDIM * HDIM * 2);
    __bf16* WtS = (__bf16*)alloc((size_t)LAYERS * HDIM * HDIM * 2);
    __bf16* WtH = (__bf16*)alloc((size_t)LAYERS * HDIM * HDIM * 2);

    // ---- one-time weight transpose+cast (tiny) ----
    {
        const int kn = HDIM * HDIM, blocks = (kn + 255) / 256;
        for (int i = 0; i < LAYERS; ++i) {
            prep_w<<<blocks, 256, 0, stream>>>(Wk  + (size_t)i*kn, WtK + (size_t)i*kn, HDIM, HDIM);
            prep_w<<<blocks, 256, 0, stream>>>(Wq  + (size_t)i*kn, WtQ + (size_t)i*kn, HDIM, HDIM);
            prep_w<<<blocks, 256, 0, stream>>>(Wv  + (size_t)i*kn, WtV + (size_t)i*kn, HDIM, HDIM);
            prep_w<<<blocks, 256, 0, stream>>>(Wsk + (size_t)i*kn, WtS + (size_t)i*kn, HDIM, HDIM);
        }
        const int knh = LAYERS * HDIM * HDIM;
        prep_w<<<(knh + 255) / 256, 256, 0, stream>>>(Wh, WtH, LAYERS * HDIM, HDIM);
    }

    // ---- input projections (fp32 A read once, bf16 out) ----
    {
        int T = (Nn / 16) * (HDIM / 16);
        wmma_gemm_f32a<__bf16><<<(T + 7) / 8, 256, 0, stream>>>(
            nf, Wn, bn, xb, Nn, IN_NODE, HDIM);
        T = (E / 16) * (EHDIM / 16);
        wmma_gemm_f32a<__bf16><<<(T + 7) / 8, 256, 0, stream>>>(
            ea, We, be, epb, E, IN_EDGE, EHDIM);
    }

    const int Tn      = (Nn / 16) * (HDIM / 16);
    const int blocksN = (Tn + 7) / 8;

    for (int i = 0; i < LAYERS; ++i) {
        const size_t wo = (size_t)i * HDIM * HDIM;
        wmma_gemm_bf16a<4, float><<<blocksN, 256, 0, stream>>>(
            xb, WtK + wo, bk + (size_t)i*HDIM, kf, Nn, HDIM);
        wmma_gemm_bf16a<4, float><<<blocksN, 256, 0, stream>>>(
            xb, WtQ + wo, bq + (size_t)i*HDIM, qf, Nn, HDIM);
        wmma_gemm_bf16a<4, float><<<blocksN, 256, 0, stream>>>(
            xb, WtV + wo, bv + (size_t)i*HDIM, vf, Nn, HDIM);
        wmma_gemm_bf16a<4, float><<<blocksN, 256, 0, stream>>>(
            xb, WtS + wo, cb + (size_t)i*HDIM, xs, Nn, HDIM);

        zero_f32<<<(int)((nhf + 255) / 256), 256, 0, stream>>>(agg, nhf);

        edge_gate_agg<<<8192, 256, 0, stream>>>(
            kf, qf, vf, epb, src, dst,
            Wed + (size_t)i * EHDIM * HDIM, bed + (size_t)i * HDIM, agg, E);

        ln_res<<<(Nn + 7) / 8, 256, 0, stream>>>(
            agg, xs, lng + (size_t)i * HDIM, lnb + (size_t)i * HDIM,
            xb, ctb, Nn, i);
    }

    // ---- final projection: out = concat(hidden) @ Wh + bh (fp32 out) ----
    wmma_gemm_bf16a<12, float><<<blocksN, 256, 0, stream>>>(
        ctb, WtH, bh, (float*)d_out, Nn, HDIM);
}
